// LSTMCell_87668872446416
// MI455X (gfx1250) — matrix operand order repaired and checked by reading, not compiled, not run on hardware
//
#include <hip/hip_runtime.h>
#include <cstdint>
#include <cstddef>

// ---------------------------------------------------------------------------
// LSTM cell, B=16384, D=H=1024.
// pre = [x|h] (Bx2048) @ [Wi|Wh]^T (2048x4096) + bi ; gates -> o,new_h,new_c.
// bf16 WMMA (v_wmma_f32_16x16x32_bf16), fp32 accumulate, fused epilogue.
// Wave tile: 64 rows x 16 cols x 4 gates (16 accumulators, each B fragment
// reused by 4 WMMAs). A staged 64x128 per stage via double-buffered LDS,
// one barrier per stage. B stream software-pipelined one K-step ahead.
// ---------------------------------------------------------------------------

typedef __attribute__((ext_vector_type(16))) __bf16 v16bf;
typedef __attribute__((ext_vector_type(8)))  float  v8f;

struct BF16Frag { uint4 lo, hi; };   // 32 bytes == one v16bf fragment
static_assert(sizeof(BF16Frag) == 32, "frag size");
static_assert(sizeof(v16bf)    == 32, "v16bf size");

static constexpr int kB   = 16384;   // batch
static constexpr int kH   = 1024;    // hidden (== D)
static constexpr int kK   = 2048;    // D + H reduction length
static constexpr int kNG  = 4096;    // 4 gates * H output columns

static constexpr int MROWS  = 64;    // rows per block (== rows per wave tile)
static constexpr int KCHUNK = 128;   // K staged per LDS buffer
static constexpr int NSTAGE = kK / KCHUNK;        // 16
static constexpr int NSTEP  = kK / 32;            // 64 WMMA K-steps total

// fp32 -> bf16 round-to-nearest-even
__device__ __forceinline__ unsigned short f2bf(float f) {
  unsigned u = __builtin_bit_cast(unsigned, f);
  u += 0x7FFFu + ((u >> 16) & 1u);
  return (unsigned short)(u >> 16);
}

__device__ __forceinline__ float sigmoidf_fast(float x) {
  return __fdividef(1.0f, 1.0f + __expf(-x));
}

// sign-safe fast tanh: one v_exp_f32 + fast divide
__device__ __forceinline__ float tanhf_fast(float x) {
  const float t = fabsf(x);
  const float e = __expf(-2.0f * t);
  const float r = __fdividef(1.0f - e, 1.0f + e);
  return copysignf(r, x);
}

// 32B fragment load (lowers to ds_load_b128 x2 or global_load_b128 x2)
__device__ __forceinline__ v16bf load_frag(const unsigned short* p) {
  BF16Frag f;
  f.lo = *(const uint4*)p;
  f.hi = *(const uint4*)(p + 8);
  return __builtin_bit_cast(v16bf, f);
}

// dst[row][0:1024) = a[row][:], dst[row][1024:2048) = b[row][:], as bf16.
__global__ __launch_bounds__(256)
void pack_pair_kernel(const float* __restrict__ a,
                      const float* __restrict__ b,
                      unsigned short* __restrict__ dst) {
  const int idx = blockIdx.x * blockDim.x + threadIdx.x;
  const int row = idx >> 9;              // 2048/4 = 512 quads per row
  const int j   = (idx & 511) << 2;
  const float* src = (j < 1024) ? (a + (size_t)row * 1024 + j)
                                : (b + (size_t)row * 1024 + (j - 1024));
  const float4 v = *(const float4*)src;
  ushort4 o;
  o.x = f2bf(v.x); o.y = f2bf(v.y); o.z = f2bf(v.z); o.w = f2bf(v.w);
  *(ushort4*)(dst + (size_t)row * kK + j) = o;
}

// Block: 256 threads = 8 waves, tile 64 rows x 128 hidden cols.
// Wave: 64 rows x 16 cols, all four gates.
__global__ __launch_bounds__(256)
void lstm_wmma_kernel(const unsigned short* __restrict__ xh,   // [16384 x 2048] bf16
                      const unsigned short* __restrict__ wt,   // [4096 x 2048] bf16
                      const float* __restrict__ bi,            // [4 x 1024]
                      const float* __restrict__ cin,           // [16384 x 1024]
                      float* __restrict__ out) {               // o | new_h | new_c
  __shared__ __align__(16) unsigned short As[2][MROWS * KCHUNK];  // 2 x 16 KB

  const int tid  = threadIdx.x;
  const int lane = tid & 31;
  const int wave = tid >> 5;
  const int rowBase = blockIdx.y * MROWS;                      // 64 rows / block
  const int colBase = (blockIdx.x << 7) + (wave << 4);         // 16 cols / wave

  // --- A-fragment LDS read base (contiguous k = 16h + s, same order as B) -------------
  // lanes 0..15: M=lane, K {0..15}; lanes 16..31: M=lane-16,
  // K {16..31}.  Fragment = 32 contiguous bytes @ kA.
  const int mA = lane & 15;
  const int kA = (lane < 16) ? 0 : 16;
  const int aOff = mA * KCHUNK + kA;                           // elements

  // --- B-fragment global pointers (32x16 B: lane = column) ----------------
  const int nB   = lane & 15;
  const int kBof = (lane < 16) ? 0 : 16;
  const int colB = colBase + nB;
  const unsigned short* wp[4];
#pragma unroll
  for (int g = 0; g < 4; ++g)
    wp[g] = wt + ((size_t)(g * kH + colB)) * kK + kBof;

  // --- cooperative A staging addresses ------------------------------------
  // 64 rows x 128 K bf16 = 16 KB; thread t stages 64 B: row=t/4, koff=(t%4)*32
  const int srow = tid >> 2;
  const int skof = (tid & 3) << 5;
  const unsigned short* xsrc = xh + (size_t)(rowBase + srow) * kK + skof;
  const int sdst = srow * KCHUNK + skof;                       // elements

  const v8f vzero = {0, 0, 0, 0, 0, 0, 0, 0};
  v8f acc[4][4];                          // [gate][row-tile], 128 VGPRs
#pragma unroll
  for (int g = 0; g < 4; ++g)
#pragma unroll
    for (int t = 0; t < 4; ++t) acc[g][t] = vzero;

  // prologue: stage 0 into buffer 0
  uint4 ga[4];
#pragma unroll
  for (int q = 0; q < 4; ++q) ga[q] = *(const uint4*)(xsrc + q * 8);
#pragma unroll
  for (int q = 0; q < 4; ++q) *(uint4*)(&As[0][sdst + q * 8]) = ga[q];
  __syncthreads();

  // prologue: B fragments for K-step 0
  v16bf bc[4];
#pragma unroll
  for (int g = 0; g < 4; ++g) bc[g] = load_frag(wp[g]);

  for (int s = 0; s < NSTAGE; ++s) {
    const int cur = s & 1;

    // issue next stage's A loads early (overlap with 64 WMMAs below)
    if (s + 1 < NSTAGE) {
#pragma unroll
      for (int q = 0; q < 4; ++q)
        ga[q] = *(const uint4*)(xsrc + (s + 1) * KCHUNK + q * 8);
#pragma unroll
      for (int g = 0; g < 4; ++g)                 // weight prefetch 1 stage ahead
        __builtin_prefetch(wp[g] + (s + 1) * KCHUNK, 0, 1);
    }

    const unsigned short* aBase = &As[cur][aOff];

#pragma unroll
    for (int kk = 0; kk < 4; ++kk) {
      const int j  = (s << 2) + kk;               // global K-step index
      const int jn = (j + 1 < NSTEP) ? (j + 1) : (NSTEP - 1);

      // software-pipelined B loads for the NEXT K-step
      v16bf bn[4];
#pragma unroll
      for (int g = 0; g < 4; ++g) bn[g] = load_frag(wp[g] + jn * 32);

      // A fragments for this K-step (row tiles 0..3 of the 64-row wave tile)
      v16bf a[4];
#pragma unroll
      for (int t = 0; t < 4; ++t)
        a[t] = load_frag(aBase + kk * 32 + t * 16 * KCHUNK);

#pragma unroll
      for (int g = 0; g < 4; ++g)
#pragma unroll
        for (int t = 0; t < 4; ++t)
          acc[g][t] = __builtin_amdgcn_wmma_f32_16x16x32_bf16(
              false, a[t], false, bc[g], (short)0, acc[g][t], false, false);

#pragma unroll
      for (int g = 0; g < 4; ++g) bc[g] = bn[g];
    }

    // store next stage into the other buffer, then one barrier
    if (s + 1 < NSTAGE) {
#pragma unroll
      for (int q = 0; q < 4; ++q)
        *(uint4*)(&As[cur ^ 1][sdst + q * 8]) = ga[q];
    }
    __syncthreads();
  }

  // --- fused LSTM epilogue ------------------------------------------------
  // C/D layout: VGPR r -> (M = r + (lane>=16 ? 8 : 0), N = lane&15)
  const int mC  = (lane >> 4) << 3;        // 0 or 8
  const int nC  = lane & 15;
  const int col = colBase + nC;

  const float b_i = bi[0 * kH + col];
  const float b_f = bi[1 * kH + col];
  const float b_g = bi[2 * kH + col];
  const float b_o = bi[3 * kH + col];

  float* o_out = out;
  float* h_out = out + (size_t)kB * kH;
  float* c_out = out + 2 * (size_t)kB * kH;

#pragma unroll
  for (int t = 0; t < 4; ++t) {
#pragma unroll
    for (int r = 0; r < 8; ++r) {
      const int row = rowBase + t * 16 + mC + r;
      const size_t idx = (size_t)row * kH + col;
      const float ig = sigmoidf_fast(acc[0][t][r] + b_i);
      const float fg = sigmoidf_fast(acc[1][t][r] + b_f);
      const float gg = tanhf_fast(acc[2][t][r] + b_g);
      const float og = sigmoidf_fast(acc[3][t][r] + b_o);
      const float cn = fg * cin[idx] + ig * gg;
      const float hn = og * tanhf_fast(cn);
      o_out[idx] = og;
      h_out[idx] = hn;
      c_out[idx] = cn;
    }
  }
}

extern "C" void kernel_launch(void* const* d_in, const int* in_sizes, int n_in,
                              void* d_out, int out_size, void* d_ws, size_t ws_size,
                              hipStream_t stream) {
  (void)in_sizes; (void)n_in; (void)out_size; (void)ws_size;
  const float* x  = (const float*)d_in[0];
  const float* h  = (const float*)d_in[1];
  const float* c  = (const float*)d_in[2];
  const float* Wi = (const float*)d_in[3];
  const float* bi = (const float*)d_in[4];
  const float* Wh = (const float*)d_in[5];

  unsigned short* xh = (unsigned short*)d_ws;               // 16384*2048 bf16 = 64 MB
  unsigned short* wt = xh + (size_t)kB * kK;                //  4096*2048 bf16 = 16 MB

  pack_pair_kernel<<<(kB * (kK / 4)) / 256, 256, 0, stream>>>(x, h, xh);
  pack_pair_kernel<<<(kNG * (kK / 4)) / 256, 256, 0, stream>>>(Wi, Wh, wt);

  dim3 grid(kH / 128, kB / MROWS);          // 8 x 256 blocks, 8 waves each
  lstm_wmma_kernel<<<grid, 256, 0, stream>>>(xh, wt, bi, c, (float*)d_out);
}
